// HourglassNet_66623532695953
// MI455X (gfx1250) — matrix operand order repaired and checked by reading, not compiled, not run on hardware
//
#include <hip/hip_runtime.h>

// ---------------------------------------------------------------------------
// CDNA5 WMMA types (wave32): bf16 A/B fragments, f32 accumulator.
// ---------------------------------------------------------------------------
typedef __attribute__((ext_vector_type(16))) __bf16 v16bf;   // 32B
typedef __attribute__((ext_vector_type(8)))  __bf16 v8bf;    // 16B
typedef __attribute__((ext_vector_type(8)))  float  v8f;

#define BNS 0.9999950000374997f  // 1/sqrt(1+1e-5): BatchNorm eval-mode scale

union U16 { v16bf v; v8bf h[2]; };

// ---------------------------------------------------------------------------
// Weight repack: OIHW fp32 -> bf16 [COp][Kp], K-order (kh*KW+kw)*CI + ci,
// CO padded to 64 (zero rows), K padded to 32 -> all A loads unconditional
// aligned 16B vector loads. One-time cost per launch.
// ---------------------------------------------------------------------------
__global__ void repack_w_k(const float* __restrict__ in, __bf16* __restrict__ out,
                           int CO, int CI, int KHW, int Kp, long n) {
  long id = (long)blockIdx.x * blockDim.x + threadIdx.x;
  if (id >= n) return;
  int k  = (int)(id % Kp);
  int co = (int)(id / Kp);
  int K  = CI * KHW;
  float v = 0.0f;
  if (co < CO && k < K) {
    int ci = k % CI;
    int p  = k / CI;
    v = in[((size_t)co * CI + ci) * KHW + p];
  }
  out[id] = (__bf16)v;
}

// Stem weight repack: (64,1,7,7) -> [64][64], k = kh*8+kw, zero-padded so the
// stem K-loop needs only shift/mask decode and no bounds checks.
__global__ void repack_stem_k(const float* __restrict__ in, __bf16* __restrict__ out) {
  int id = blockIdx.x * blockDim.x + threadIdx.x;   // 64*64
  if (id >= 64 * 64) return;
  int k  = id & 63;
  int co = id >> 6;
  int kh = k >> 3, kw = k & 7;
  float v = (kh < 7 && kw < 7) ? in[co * 49 + kh * 7 + kw] : 0.0f;
  out[id] = (__bf16)v;
}

// ---------------------------------------------------------------------------
// Activation staging: NCHW fp32 -> halo-padded NHWC bf16 with fused
// pre-activation transform v -> (relu?)(v)*(g*BNS)+b, halo ring = 0.
// One block per (n, padded row); threads stride channels; coalesced writes;
// only ONE integer divide per block.  Output dims: (H+2P) x (W+2P) x C.
// ---------------------------------------------------------------------------
__global__ void tr_nhwc_k(const float* __restrict__ x, __bf16* __restrict__ y,
                          int C, int H, int W, int P,
                          const float* __restrict__ g, const float* __restrict__ b,
                          int relu) {
  const int Hp = H + 2 * P, Wpad = W + 2 * P;
  const int row = blockIdx.x;            // n*Hp + hp
  const int hp  = row % Hp;
  const int n   = row / Hp;
  const int h   = hp - P;
  const bool rowin = (h >= 0 && h < H);
  const float* xr = x + (size_t)n * C * H * W + (size_t)(rowin ? h : 0) * W;
  __bf16* yr = y + (size_t)row * Wpad * C;
  for (int wp = 0; wp < Wpad; ++wp) {
    const int w = wp - P;
    const bool in = rowin && w >= 0 && w < W;   // uniform across threads
    for (int c = threadIdx.x; c < C; c += 256) {
      float v = 0.0f;
      if (in) {
        v = xr[(size_t)c * H * W + w];
        if (relu) v = fmaxf(v, 0.0f);
        if (g) v = v * (g[c] * BNS) + b[c];
      }
      yr[(size_t)wp * C + c] = (__bf16)v;
    }
  }
}

// ---------------------------------------------------------------------------
// Implicit-GEMM conv, templated KHxKW, halo-padded NHWC bf16 input (P=1 for
// 3x3, P=0 for 1x1), NCHW f32 output.
// One wave = 64 output channels x 32 pixels:
//   per 32-K step: 8x 16B A-loads + 2x 32B B-loads + 8x WMMA, all
//   unconditional (padded weights + halo-padded activations).
// A lane layout (16-bit A 16x32): lanes 0-15 K=0..15, lanes 16-31
// K=16..31 (contiguous, as B).  B: lanes 0-15 K=0..15, lanes 16-31 K=16..31.
// ups=1 (1x1 only): nearest-2x upsample fused into input indexing.
// Tile early-exit wave-uniform -> EXEC all-ones at every WMMA.
// ---------------------------------------------------------------------------
template <int KH, int KW>
__global__ __launch_bounds__(256) void convNxN_wmma_k(
    const __bf16* __restrict__ XB, const __bf16* __restrict__ Wq, float* __restrict__ Y,
    const float* __restrict__ bias, const float* __restrict__ skip, int out_relu,
    int N, int CI, int H, int W, int CO, int stride, int OH, int OW, int ups)
{
  constexpr int P = (KH == 3) ? 1 : 0;
  const int Hp = H + 2 * P, Wpad = W + 2 * P;
  const int lane = threadIdx.x & 31;
  const int wv   = threadIdx.x >> 5;
  const int m0   = blockIdx.y * 64;
  const long npix = (long)N * OH * OW;
  const int ntiles = (int)((npix + 31) >> 5);
  const int nt = blockIdx.x * 8 + wv;
  if (nt >= ntiles) return;          // wave-uniform
  const int n0  = nt * 32;
  const int l15 = lane & 15;
  const bool hi = lane >= 16;
  const int kA  = hi ? 16 : 0;
  const int kB  = hi ? 16 : 0;
  const int K   = KH * KW * CI;      // multiple of 32 (CI%32==0)

  // two B columns (pixels) per lane
  const __bf16* bb[2];
  size_t ob[2];
  bool nvq[2];
#pragma unroll
  for (int q = 0; q < 2; ++q) {
    long nc = n0 + q * 16 + l15;
    bool nv = nc < npix;
    long pix = nv ? nc : 0;
    int ow  = (int)(pix % OW); pix /= OW;
    int oh  = (int)(pix % OH);
    int img = (int)(pix / OH);
    int ihb, iwb;
    if (KH == 1) { ihb = ups ? (oh >> 1) : oh * stride; iwb = ups ? (ow >> 1) : ow * stride; }
    else         { ihb = oh * stride;                   iwb = ow * stride; }  // halo absorbs pad
    bb[q]  = XB + ((size_t)(img * Hp + ihb) * Wpad + iwb) * CI + kB;
    ob[q]  = (size_t)img * CO * OH * OW + (size_t)oh * OW + ow;
    nvq[q] = nv;
  }

  const __bf16* a0 = Wq + (size_t)(m0 + l15) * K + kA;  // rows padded -> valid

  v8f acc[8] = {};

#pragma unroll
  for (int p = 0; p < KH * KW; ++p) {
    const size_t boff = ((size_t)(p / KW) * Wpad + (p % KW)) * CI;
    const int aoff = p * CI;
    for (int cb = 0; cb < CI; cb += 32) {
      v16bf b0 = *(const v16bf*)(bb[0] + boff + cb);
      v16bf b1 = *(const v16bf*)(bb[1] + boff + cb);
      U16 a[4];
#pragma unroll
      for (int t = 0; t < 4; ++t) {
        const __bf16* ap = a0 + (size_t)(16 * t) * K + aoff + cb;
        a[t].h[0] = *(const v8bf*)(ap);
        a[t].h[1] = *(const v8bf*)(ap + 8);
      }
#pragma unroll
      for (int t = 0; t < 4; ++t) {
        acc[t]     = __builtin_amdgcn_wmma_f32_16x16x32_bf16(false, a[t].v, false, b0, (short)0, acc[t],     false, false);
        acc[4 + t] = __builtin_amdgcn_wmma_f32_16x16x32_bf16(false, a[t].v, false, b1, (short)0, acc[4 + t], false, false);
      }
    }
  }

  // C/D layout: lane L -> column L&15; VGPR r -> row r (+8 for hi lanes)
  const int mbL = hi ? 8 : 0;
#pragma unroll
  for (int q = 0; q < 2; ++q) {
#pragma unroll
    for (int t = 0; t < 4; ++t) {
#pragma unroll
      for (int r = 0; r < 8; ++r) {
        int mm = m0 + 16 * t + mbL + r;
        if (!nvq[q] || mm >= CO) continue;
        float v = acc[q * 4 + t][r];
        if (bias) v += bias[mm];
        size_t oi = ob[q] + (size_t)mm * OH * OW;
        if (skip) v += skip[oi];
        if (out_relu) v = fmaxf(v, 0.0f);
        Y[oi] = v;
      }
    }
  }
}

// ---------------------------------------------------------------------------
// Stem conv: 7x7 s2, CI=1, CO=64, input halo-padded by 4 (0.3% of FLOPs).
// Weights packed k=kh*8+kw (zero-padded) -> shift/mask decode and fully
// UNCONDITIONAL gathers: k with kh==7 or kw==7 reads finite halo/data values that
// multiply zero weight rows (exactly 0 contribution).  Window origin in
// padded coords: oh*2 + (P-3) = oh*2 + 1.
// ---------------------------------------------------------------------------
__global__ __launch_bounds__(256) void conv_stem_k(
    const __bf16* __restrict__ XB, const __bf16* __restrict__ Wq, float* __restrict__ Y,
    const float* __restrict__ bias, const float* __restrict__ og, const float* __restrict__ ob,
    int N, int H, int W, int OH, int OW)
{
  const int Hp = H + 8, Wpad = W + 8;
  const int lane = threadIdx.x & 31;
  const int wv   = threadIdx.x >> 5;
  const long npix = (long)N * OH * OW;
  const int ntiles = (int)((npix + 15) >> 4);
  const int nt = blockIdx.x * 8 + wv;
  if (nt >= ntiles) return;
  const int n0  = nt * 16;
  const int l15 = lane & 15;
  const bool hi = lane >= 16;
  const int kA  = hi ? 16 : 0;
  const int kB  = hi ? 16 : 0;

  const long nc = n0 + l15;
  const bool nv = nc < npix;
  long pix = nv ? nc : 0;
  const int ow  = (int)(pix % OW); pix /= OW;
  const int oh  = (int)(pix % OH);
  const int img = (int)(pix / OH);
  const __bf16* xim = XB + (size_t)(img * Hp + oh * 2 + 1) * Wpad + (ow * 2 + 1);

  v8f acc[4] = {};
#pragma unroll
  for (int k0 = 0; k0 < 64; k0 += 32) {
    v16bf b;
#pragma unroll
    for (int j = 0; j < 16; ++j) {
      int k = k0 + kB + j;
      b[j] = xim[(size_t)(k >> 3) * Wpad + (k & 7)];   // unconditional
    }
    U16 a[4];
#pragma unroll
    for (int t = 0; t < 4; ++t) {
      const __bf16* ap = Wq + (size_t)(16 * t + l15) * 64 + kA + k0;
      a[t].h[0] = *(const v8bf*)(ap);
      a[t].h[1] = *(const v8bf*)(ap + 8);
    }
#pragma unroll
    for (int t = 0; t < 4; ++t)
      acc[t] = __builtin_amdgcn_wmma_f32_16x16x32_bf16(false, a[t].v, false, b, (short)0, acc[t], false, false);
  }

  const int mbL = hi ? 8 : 0;
  const size_t obase = (size_t)img * 64 * OH * OW + (size_t)oh * OW + ow;
#pragma unroll
  for (int t = 0; t < 4; ++t) {
#pragma unroll
    for (int r = 0; r < 8; ++r) {
      int mm = 16 * t + mbL + r;
      if (!nv) continue;
      float v = acc[t][r];
      v += bias[mm];
      v = v * (og[mm] * BNS) + ob[mm];
      Y[obase + (size_t)mm * OH * OW] = fmaxf(v, 0.0f);
    }
  }
}

// ---------------------------------------------------------------------------
// Elementwise / scan helpers
// ---------------------------------------------------------------------------
__global__ void maxpool2_k(const float* __restrict__ x, float* __restrict__ y,
                           long NC, int H, int W) {
  int OH = H >> 1, OW = W >> 1;
  long n  = NC * OH * OW;
  long id = (long)blockIdx.x * blockDim.x + threadIdx.x;
  if (id >= n) return;
  int ow = (int)(id % OW); long t = id / OW;
  int oh = (int)(t % OH);  long nc = t / OH;
  const float* xp = x + nc * (long)H * W;
  long r0 = (long)(2 * oh) * W + 2 * ow;
  y[id] = fmaxf(fmaxf(xp[r0], xp[r0 + 1]), fmaxf(xp[r0 + W], xp[r0 + W + 1]));
}

__global__ void up2_k(const float* __restrict__ x, float* __restrict__ y,
                      long NC, int H, int W) {
  int OH = H * 2, OW = W * 2;
  long n  = NC * OH * OW;
  long id = (long)blockIdx.x * blockDim.x + threadIdx.x;
  if (id >= n) return;
  int ow = (int)(id % OW); long t = id / OW;
  int oh = (int)(t % OH);  long nc = t / OH;
  y[id] = x[nc * (long)H * W + (long)(oh >> 1) * W + (ow >> 1)];
}

__global__ void cummax_k(const float* __restrict__ x, float* __restrict__ y,
                         int N, int C, int H, int W, int axis, int rev) {
  long lines = (long)N * C * (axis == 2 ? W : H);
  long id = (long)blockIdx.x * blockDim.x + threadIdx.x;
  if (id >= lines) return;
  if (axis == 2) {
    int  w  = (int)(id % W);
    long nc = id / W;
    const float* xs = x + nc * (long)H * W + w;
    float*       ys = y + nc * (long)H * W + w;
    float m = -3.402823466e38f;
    if (!rev) for (int h = 0; h < H; ++h)      { m = fmaxf(m, xs[(long)h * W]); ys[(long)h * W] = m; }
    else      for (int h = H - 1; h >= 0; --h) { m = fmaxf(m, xs[(long)h * W]); ys[(long)h * W] = m; }
  } else {
    int  h  = (int)(id % H);
    long nc = id / H;
    const float* xs = x + (nc * H + h) * (long)W;
    float*       ys = y + (nc * H + h) * (long)W;
    float m = -3.402823466e38f;
    if (!rev) for (int w = 0; w < W; ++w)      { m = fmaxf(m, xs[w]); ys[w] = m; }
    else      for (int w = W - 1; w >= 0; --w) { m = fmaxf(m, xs[w]); ys[w] = m; }
  }
}

__global__ void add_k(const float* __restrict__ a, const float* __restrict__ b,
                      float* __restrict__ y, long n) {
  long i = (long)blockIdx.x * blockDim.x + threadIdx.x;
  if (i < n) y[i] = a[i] + b[i];
}

// ---------------------------------------------------------------------------
// Host orchestration.  Param flattening (dict insertion order):
//   0:x 1:conv1_w 2:conv1_b 3:bn1_g 4:bn1_b
//   5..16:layer1{g1,b1,w1,g2,b2,w2,g3,b3,w3,gs,bs,ws} 17:layer2 29:layer3
//   41..100:down0..down4  101..104:up5,up4,up3,up2
//   105..128: heads {w3,b3,w1,b1} x6
// ---------------------------------------------------------------------------
extern "C" void kernel_launch(void* const* d_in, const int* in_sizes, int n_in,
                              void* d_out, int out_size, void* d_ws, size_t ws_size,
                              hipStream_t stream) {
  (void)in_sizes; (void)n_in; (void)out_size; (void)ws_size;

  auto f = [&](int i) -> const float* { return (const float*)d_in[i]; };

  char*  base = (char*)d_ws;
  size_t off  = 0;
  auto alloc = [&](size_t bytes) -> void* {
    void* p = base + off;
    off = (off + bytes + 255) & ~(size_t)255;
    return p;
  };

  auto packw = [&](int idx, int CO, int CI, int KHW) -> const __bf16* {
    int COp = (CO + 63) & ~63;
    int Kp  = (CI * KHW + 31) & ~31;
    long n  = (long)COp * Kp;
    __bf16* dst = (__bf16*)alloc((size_t)n * 2);
    repack_w_k<<<(int)((n + 255) / 256), 256, 0, stream>>>(f(idx), dst, CO, CI, KHW, Kp, n);
    return dst;
  };

  auto trans = [&](const float* x, __bf16* y, int N, int C, int H, int W, int P,
                   const float* g, const float* b, int relu) {
    tr_nhwc_k<<<N * (H + 2 * P), 256, 0, stream>>>(x, y, C, H, W, P, g, b, relu);
  };

  auto conv1x1 = [&](const __bf16* XB, const __bf16* Wq, float* Y,
                     int N, int CI, int H, int W, int CO, int stride, int OH, int OW, int ups,
                     const float* bias, const float* skip, int relu) {
    long npix = (long)N * OH * OW;
    int ntiles = (int)((npix + 31) / 32);
    dim3 g((ntiles + 7) / 8, (CO + 63) / 64);
    convNxN_wmma_k<1, 1><<<g, 256, 0, stream>>>(XB, Wq, Y, bias, skip, relu,
                                                N, CI, H, W, CO, stride, OH, OW, ups);
  };
  auto conv3x3 = [&](const __bf16* XB, const __bf16* Wq, float* Y,
                     int N, int CI, int H, int W, int CO, int stride, int OH, int OW,
                     const float* bias, int relu) {
    long npix = (long)N * OH * OW;
    int ntiles = (int)((npix + 31) / 32);
    dim3 g((ntiles + 7) / 8, (CO + 63) / 64);
    convNxN_wmma_k<3, 3><<<g, 256, 0, stream>>>(XB, Wq, Y, bias, nullptr, relu,
                                                N, CI, H, W, CO, stride, OH, OW, 0);
  };

  struct ResW { const float *g1,*b1,*g2,*b2,*g3,*b3,*gs,*bs;
                const __bf16 *w1,*w2,*w3,*ws; int ci, co; };
  auto loadres = [&](int b, int ci, int co) -> ResW {
    ResW r; r.ci = ci; r.co = co;
    r.g1 = f(b+0);  r.b1 = f(b+1);  r.w1 = packw(b+2,  co, ci, 1);
    r.g2 = f(b+3);  r.b2 = f(b+4);  r.w2 = packw(b+5,  co, co, 9);
    r.g3 = f(b+6);  r.b3 = f(b+7);  r.w3 = packw(b+8,  co, co, 1);
    r.gs = f(b+9);  r.bs = f(b+10); r.ws = packw(b+11, co, ci, 1);
    return r;
  };

  // ---- weights
  __bf16* c1w = (__bf16*)alloc(64 * 64 * 2);
  repack_stem_k<<<16, 256, 0, stream>>>(f(1), c1w);
  const float *c1b = f(2), *bn1g = f(3), *bn1b = f(4);
  ResW L1 = loadres(5, 64, 128),  L2 = loadres(17, 128, 128), L3 = loadres(29, 128, 256);
  ResW D0 = loadres(41, 256, 256), D1 = loadres(53, 256, 384), D2 = loadres(65, 384, 384);
  ResW D3 = loadres(77, 384, 384), D4 = loadres(89, 384, 512);
  const __bf16* up5w = packw(101, 384, 512, 1);
  const __bf16* up4w = packw(102, 384, 384, 1);
  const __bf16* up3w = packw(103, 384, 384, 1);
  const __bf16* up2w = packw(104, 256, 384, 1);
  struct HeadW { const __bf16 *w3, *w1; const float *b3, *b1; int cout; };
  auto loadhead = [&](int b, int cout) -> HeadW {
    HeadW h; h.w3 = packw(b, 256, 256, 9); h.b3 = f(b + 1);
    h.w1 = packw(b + 2, cout, 256, 1); h.b1 = f(b + 3); h.cout = cout; return h;
  };
  HeadW Htlh = loadhead(105, 2), Hbrh = loadhead(109, 2), Htlt = loadhead(113, 1);
  HeadW Hbrt = loadhead(117, 1), Htlr = loadhead(121, 2), Hbrr = loadhead(125, 2);

  // ---- activation slots: five rotating f32 slots + one bf16 staging slot
  const size_t SLOT = (size_t)2 * 128 * 256 * 256;          // 16.78M f32
  const size_t BFEL = (size_t)2 * 258 * 258 * 128;          // halo-padded max
  float* S0 = (float*)alloc(SLOT * 4);
  float* S1 = (float*)alloc(SLOT * 4);
  float* S2 = (float*)alloc(SLOT * 4);
  float* S3 = (float*)alloc(SLOT * 4);
  float* S4 = (float*)alloc(SLOT * 4);
  __bf16* BF = (__bf16*)alloc(BFEL * 2);
  float* d0b = (float*)alloc((size_t)2*256*64*64*4);
  float* d1b = (float*)alloc((size_t)2*384*32*32*4);
  float* d2b = (float*)alloc((size_t)2*384*16*16*4);
  float* d3b = (float*)alloc((size_t)2*384*8*8*4);
  float* d4b = (float*)alloc((size_t)2*512*4*4*4);
  float* u3b = (float*)alloc((size_t)2*384*8*8*4);
  float* u2b = (float*)alloc((size_t)2*384*16*16*4);
  float* u1b = (float*)alloc((size_t)2*384*32*32*4);
  float* u0b = (float*)alloc((size_t)2*256*64*64*4);

  // pre-activation bottleneck residual
  auto runres = [&](const ResW& P, const float* X, int H, int W, int stride,
                    float* t1, float* t2, float* s, float* out) {
    int OH = (H - 1) / stride + 1, OW = (W - 1) / stride + 1;
    trans(X, BF, 2, P.ci, H, W, 0, P.gs, P.bs, 1);
    conv1x1(BF, P.ws, s,  2, P.ci, H, W, P.co, stride, OH, OW, 0, nullptr, nullptr, 0);
    trans(X, BF, 2, P.ci, H, W, 0, P.g1, P.b1, 1);
    conv1x1(BF, P.w1, t1, 2, P.ci, H, W, P.co, 1, H, W, 0, nullptr, nullptr, 0);
    trans(t1, BF, 2, P.co, H, W, 1, P.g2, P.b2, 1);
    conv3x3(BF, P.w2, t2, 2, P.co, H, W, P.co, stride, OH, OW, nullptr, 0);
    trans(t2, BF, 2, P.co, OH, OW, 0, P.g3, P.b3, 1);
    conv1x1(BF, P.w3, out, 2, P.co, OH, OW, P.co, 1, OH, OW, 0, nullptr, s, 0);
  };

  const float* x = f(0);

  // stem: 7x7 s2 p3 (halo P=4) + bias + BN + ReLU -> S0 (2x64x256x256)
  trans(x, BF, 2, 1, 512, 512, 4, nullptr, nullptr, 0);
  { long npix = (long)2 * 256 * 256;
    int ntiles = (int)((npix + 15) / 16);
    dim3 g((ntiles + 7) / 8, 1);
    conv_stem_k<<<g, 256, 0, stream>>>(BF, c1w, S0, c1b, bn1g, bn1b,
                                       2, 512, 512, 256, 256); }

  runres(L1, S0, 256, 256, 1, S2, S3, S1, S4);                 // 2x128x256x256
  { long n = (long)2*128*128*128;
    maxpool2_k<<<(int)((n+255)/256),256,0,stream>>>(S4, S0, (long)2*128, 256, 256); }
  runres(L2, S0, 128, 128, 1, S2, S3, S1, S4);                 // 2x128x128x128
  runres(L3, S4, 128, 128, 1, S2, S3, S1, S0);                 // 2x256x128x128

  // hourglass down
  runres(D0, S0,  128, 128, 2, S2, S3, S1, d0b);               // 2x256x64x64
  runres(D1, d0b,  64,  64, 2, S2, S3, S1, d1b);               // 2x384x32x32
  runres(D2, d1b,  32,  32, 2, S2, S3, S1, d2b);               // 2x384x16x16
  runres(D3, d2b,  16,  16, 2, S2, S3, S1, d3b);               // 2x384x8x8
  runres(D4, d3b,   8,   8, 2, S2, S3, S1, d4b);               // 2x512x4x4

  // hourglass up: 1x1 conv with fused nearest-2x upsample + skip add
  trans(d4b, BF, 2, 512, 4, 4, 0, nullptr, nullptr, 0);
  conv1x1(BF, up5w, u3b, 2, 512, 4, 4, 384, 1,  8,  8, 1, nullptr, d3b, 0);
  trans(u3b, BF, 2, 384, 8, 8, 0, nullptr, nullptr, 0);
  conv1x1(BF, up4w, u2b, 2, 384, 8, 8, 384, 1, 16, 16, 1, nullptr, d2b, 0);
  trans(u2b, BF, 2, 384, 16, 16, 0, nullptr, nullptr, 0);
  conv1x1(BF, up3w, u1b, 2, 384, 16, 16, 384, 1, 32, 32, 1, nullptr, d1b, 0);
  trans(u1b, BF, 2, 384, 32, 32, 0, nullptr, nullptr, 0);
  conv1x1(BF, up2w, u0b, 2, 384, 32, 32, 256, 1, 64, 64, 1, nullptr, d0b, 0);
  { long n = (long)2*256*128*128;
    up2_k<<<(int)((n+255)/256),256,0,stream>>>(u0b, S0, (long)2*256, 64, 64); }  // hh -> S0

  // corner pooling: tl = revcummaxH + revcummaxW ; br = cummaxH + cummaxW
  const long NP = (long)2*256*128*128;
  auto cmx = [&](const float* xi, float* yo, int axis, int rev) {
    long lines = (long)2*256*128;
    cummax_k<<<(int)((lines+255)/256),256,0,stream>>>(xi, yo, 2, 256, 128, 128, axis, rev);
  };
  cmx(S0, S1, 2, 1); cmx(S0, S2, 3, 1);
  add_k<<<(int)((NP+255)/256),256,0,stream>>>(S1, S2, S3, NP);  // tl -> S3
  cmx(S0, S1, 2, 0); cmx(S0, S2, 3, 0);
  add_k<<<(int)((NP+255)/256),256,0,stream>>>(S1, S2, S4, NP);  // br -> S4

  // heads: 3x3(+bias,ReLU) then 1x1(+bias) into d_out slices
  float* out = (float*)d_out;
  auto runhead = [&](const HeadW& Hd, const float* src, float* tmp, float* dst) {
    trans(src, BF, 2, 256, 128, 128, 1, nullptr, nullptr, 0);
    conv3x3(BF, Hd.w3, tmp, 2, 256, 128, 128, 256, 1, 128, 128, Hd.b3, 1);
    trans(tmp, BF, 2, 256, 128, 128, 0, nullptr, nullptr, 0);
    conv1x1(BF, Hd.w1, dst, 2, 256, 128, 128, Hd.cout, 1, 128, 128, 0, Hd.b1, nullptr, 0);
  };
  runhead(Htlh, S3, S0, out + 0);        // tl_heats (2ch)
  runhead(Hbrh, S4, S0, out + 65536);    // br_heats (2ch)
  runhead(Htlt, S3, S0, out + 131072);   // tl_tags  (1ch)
  runhead(Hbrt, S4, S0, out + 163840);   // br_tags  (1ch)
  runhead(Htlr, S3, S0, out + 196608);   // tl_regrs (2ch)
  runhead(Hbrr, S4, S0, out + 262144);   // br_regrs (2ch)
}
